// GroupedQueryAttention_46548855554139
// MI455X (gfx1250) — matrix operand order repaired and checked by reading, not compiled, not run on hardware
//
#include <hip/hip_runtime.h>
#include <math.h>

// Problem constants (match reference)
#define BSZ   2
#define SLEN  2048
#define DMOD  2048
#define NH    16
#define NKVH  8
#define HDIM  128
// GROUPS = NH / NKVH = 2

typedef __attribute__((ext_vector_type(16))) __bf16 v16bf;
typedef __attribute__((ext_vector_type(8)))  float  v8f;

union Frag {
  v16bf f;
  uint4 u[2];
  unsigned int w[8];
};

__device__ __forceinline__ unsigned short f2bf(float x) {
  unsigned int u = __float_as_uint(x);
  u += 0x7FFFu + ((u >> 16) & 1u);   // round-to-nearest-even
  return (unsigned short)(u >> 16);
}
__device__ __forceinline__ unsigned int pack2bf(float lo, float hi) {
  return (unsigned int)f2bf(lo) | ((unsigned int)f2bf(hi) << 16);
}

// A-operand (16-bit, 16x32 MxK), this code's contiguous k order: VGPR v: K = half*16 + 2v + {0,1}
// -> two b128 loads from row-major [m][k] at k0+half*16 (contiguous 16 elems).
__device__ __forceinline__ void load_a_frag(Frag& fr, const unsigned short* __restrict__ base,
                                            int stride, int k0) {
  const int lane = threadIdx.x & 31;
  const int r = lane & 15, half = lane >> 4;
  const unsigned short* p = base + (size_t)r * stride + k0 + half * 16;
  fr.u[0] = *(const uint4*)(p);
  fr.u[1] = *(const uint4*)(p + 8);
}

// B-operand (16-bit, 32x16 KxN), this code's contiguous k order: VGPR v: K = half*16 + 2v + {0,1}
// -> two b128 loads from row-major [n][k] at k0+half*16 (contiguous 16 elems).
__device__ __forceinline__ void load_b_frag(Frag& fr, const unsigned short* __restrict__ base,
                                            int stride, int k0) {
  const int lane = threadIdx.x & 31;
  const int r = lane & 15, half = lane >> 4;
  const unsigned short* p = base + (size_t)r * stride + k0 + half * 16;
  fr.u[0] = *(const uint4*)(p);
  fr.u[1] = *(const uint4*)(p + 8);
}

__device__ __forceinline__ v8f wmma_bf16(const Frag& a, const Frag& b, v8f c) {
  return __builtin_amdgcn_wmma_f32_16x16x32_bf16(false, a.f, false, b.f,
                                                 (short)0, c, false, false);
}

// ---------------------------------------------------------------------------
// f32 -> bf16 converter (2 elements / thread, dword stores)
// ---------------------------------------------------------------------------
__global__ __launch_bounds__(256)
void cvt_f32_to_bf16(const float* __restrict__ s, unsigned int* __restrict__ d, int n2) {
  int i = blockIdx.x * 256 + threadIdx.x;
  if (i < n2) d[i] = pack2bf(s[2 * i], s[2 * i + 1]);
}

// ---------------------------------------------------------------------------
// Fused QKV projection + per-head RMSNorm + RoPE (+ V transpose store)
// One wave computes a 16-token x 128-col (one head) tile. Register-only epilogue.
// grid = (SLEN/16, NH+NKVH+NKVH, BSZ), block = 32
// ---------------------------------------------------------------------------
__global__ __launch_bounds__(32)
void qkv_proj_kernel(const unsigned short* __restrict__ Xb,
                     const unsigned short* __restrict__ Wqb,
                     const unsigned short* __restrict__ Wkb,
                     const unsigned short* __restrict__ Wvb,
                     const float* __restrict__ qnw,
                     const float* __restrict__ knw,
                     unsigned short* __restrict__ Q,   // [B][NH][S][HD]
                     unsigned short* __restrict__ K,   // [B][NKVH][S][HD]
                     unsigned short* __restrict__ Vt)  // [B][NKVH][HD][S]
{
  const int m0   = blockIdx.x * 16;
  const int slot = blockIdx.y;           // 0..15 Q, 16..23 K, 24..31 V
  const int b    = blockIdx.z;

  int kind, head;
  const unsigned short* wbase;
  if (slot < NH)              { kind = 0; head = slot;        wbase = Wqb + (size_t)head * HDIM * DMOD; }
  else if (slot < NH + NKVH)  { kind = 1; head = slot - NH;   wbase = Wkb + (size_t)head * HDIM * DMOD; }
  else                        { kind = 2; head = slot - NH - NKVH; wbase = Wvb + (size_t)head * HDIM * DMOD; }

  const unsigned short* abase = Xb + ((size_t)b * SLEN + m0) * DMOD;

  v8f acc[8];
  {
    v8f z = {};
    #pragma unroll
    for (int j = 0; j < 8; ++j) acc[j] = z;
  }

  for (int kk = 0; kk < DMOD; kk += 32) {
    Frag a;
    load_a_frag(a, abase, DMOD, kk);
    #pragma unroll
    for (int j = 0; j < 8; ++j) {
      Frag bf;
      load_b_frag(bf, wbase + (size_t)(j * 16) * DMOD, DMOD, kk);
      acc[j] = wmma_bf16(a, bf, acc[j]);
    }
  }

  const int lane = threadIdx.x & 31;
  const int n = lane & 15, half = lane >> 4;

  if (kind == 2) {
    // V: no norm/rope, store transposed [hd][s] for flash A-operand loads
    unsigned short* vb = Vt + (size_t)(b * NKVH + head) * HDIM * SLEN;
    #pragma unroll
    for (int j = 0; j < 8; ++j) {
      int col = j * 16 + n;
      #pragma unroll
      for (int r = 0; r < 8; ++r) {
        int row = half * 8 + r;
        vb[(size_t)col * SLEN + (m0 + row)] = f2bf(acc[j][r]);
      }
    }
    return;
  }

  // RMSNorm: row (token) sum of squares across 128 cols; rows of one half live
  // in that half's 16 lanes -> shfl_xor(1,2,4,8) reduction stays in-half.
  float rrms[8];
  #pragma unroll
  for (int r = 0; r < 8; ++r) {
    float s = 0.f;
    #pragma unroll
    for (int j = 0; j < 8; ++j) s += acc[j][r] * acc[j][r];
    s += __shfl_xor(s, 1, 32);
    s += __shfl_xor(s, 2, 32);
    s += __shfl_xor(s, 4, 32);
    s += __shfl_xor(s, 8, 32);
    rrms[r] = rsqrtf(s * (1.0f / 128.0f) + 1e-6f);
  }

  const float* nw = (kind == 0) ? qnw : knw;
  unsigned short* obase = (kind == 0)
      ? Q + ((size_t)(b * NH   + head) * SLEN + m0) * HDIM
      : K + ((size_t)(b * NKVH + head) * SLEN + m0) * HDIM;

  const float LOG_BASE = 9.2103403719761836f;  // ln(10000)
  #pragma unroll
  for (int j = 0; j < 8; ++j) {
    const int col  = j * 16 + n;
    const int jp   = j ^ 4;                    // pair col = col ^ 64 lives in tile j^4, same lane
    const float wc = nw[col];
    const float wp = nw[col ^ 64];
    const float sgn = (col < 64) ? -1.f : 1.f;
    const float invf = expf(-(float)(col & 63) * (LOG_BASE / 64.f));
    #pragma unroll
    for (int r = 0; r < 8; ++r) {
      const int row = half * 8 + r;
      const float g  = rrms[r];
      const float xn = acc[j][r]  * g * wc;
      const float xp = acc[jp][r] * g * wp;
      float sv, cv;
      sincosf((float)(m0 + row) * invf, &sv, &cv);
      obase[(size_t)row * HDIM + col] = f2bf(xn * cv + sgn * xp * sv);
    }
  }
}

// ---------------------------------------------------------------------------
// Causal flash attention, transposed tiles: S^T = K*Q^T, O^T = V^T * P^T.
// One wave owns 16 queries; per-query softmax stats are per-lane.
// grid = (SLEN/64, NH, BSZ), block = 128 (4 independent waves)
// ---------------------------------------------------------------------------
__global__ __launch_bounds__(128)
void flash_attn_kernel(const unsigned short* __restrict__ Q,
                       const unsigned short* __restrict__ K,
                       const unsigned short* __restrict__ Vt,
                       unsigned short* __restrict__ AO)  // [B][S][NH*HD]
{
  const int wave = threadIdx.x >> 5;
  const int lane = threadIdx.x & 31;
  const int n = lane & 15, half = lane >> 4;
  const int b  = blockIdx.z;
  const int h  = blockIdx.y;
  const int q0 = blockIdx.x * 64 + wave * 16;
  const int kvh = h >> 1;  // GROUPS = 2

  const unsigned short* qbase = Q  + ((size_t)(b * NH   + h)   * SLEN + q0) * HDIM;
  const unsigned short* kbase = K  +  (size_t)(b * NKVH + kvh) * SLEN * HDIM;
  const unsigned short* vbase = Vt +  (size_t)(b * NKVH + kvh) * HDIM * SLEN;

  Frag qf[4];  // Q as B-operand: loads from row-major [query][hd]
  #pragma unroll
  for (int d = 0; d < 4; ++d) load_b_frag(qf[d], qbase, HDIM, d * 32);

  v8f acc[8];
  {
    v8f z = {};
    #pragma unroll
    for (int t = 0; t < 8; ++t) acc[t] = z;
  }
  float m_i = -1e30f, l_i = 0.f;
  const float scale = 0.08838834764831845f;  // 1/sqrt(128)
  const int qg = q0 + n;                     // this lane's query position

  for (int kc = 0; kc < q0 + 16; kc += 32) {
    v8f s0 = {}, s1 = {};
    #pragma unroll
    for (int d = 0; d < 4; ++d) {
      Frag a0, a1;  // K rows as A-operand
      load_a_frag(a0, kbase + (size_t)kc * HDIM,        HDIM, d * 32);
      load_a_frag(a1, kbase + (size_t)(kc + 16) * HDIM, HDIM, d * 32);
      s0 = wmma_bf16(a0, qf[d], s0);
      s1 = wmma_bf16(a1, qf[d], s1);
    }

    // scale + causal mask (rows = keys, cols = queries)
    float p0[8], p1[8];
    float cm = -1e30f;
    #pragma unroll
    for (int r = 0; r < 8; ++r) {
      const int key0 = kc + half * 8 + r;
      const float v0 = (key0      <= qg) ? s0[r] * scale : -1e30f;
      const float v1 = (key0 + 16 <= qg) ? s1[r] * scale : -1e30f;
      p0[r] = v0; p1[r] = v1;
      cm = fmaxf(cm, fmaxf(v0, v1));
    }
    cm = fmaxf(cm, __shfl_xor(cm, 16, 32));   // combine the two halves of each query column
    const float m_new = fmaxf(m_i, cm);
    const float alpha = __expf(m_i - m_new);
    float rs = 0.f;
    #pragma unroll
    for (int r = 0; r < 8; ++r) {
      p0[r] = __expf(p0[r] - m_new);
      p1[r] = __expf(p1[r] - m_new);
      rs += p0[r] + p1[r];
    }
    rs += __shfl_xor(rs, 16, 32);
    l_i = l_i * alpha + rs;
    m_i = m_new;
    #pragma unroll
    for (int t = 0; t < 8; ++t)
      #pragma unroll
      for (int r = 0; r < 8; ++r) acc[t][r] *= alpha;

    // Pack P^T into a B-operand: lane-half h needs keys h*16..h*16+15 of its
    // query column; rows 0-7 of a tile live in half 0, rows 8-15 in half 1.
    float lo[8], hi[8];
    #pragma unroll
    for (int r = 0; r < 8; ++r) {
      const float x0 = __shfl_xor(p0[r], 16, 32);
      const float x1 = __shfl_xor(p1[r], 16, 32);
      lo[r] = (half == 0) ? p0[r] : x1;
      hi[r] = (half == 0) ? x0    : p1[r];
    }
    Frag pf;
    #pragma unroll
    for (int v = 0; v < 4; ++v) {
      pf.w[v]     = pack2bf(lo[2 * v], lo[2 * v + 1]);
      pf.w[v + 4] = pack2bf(hi[2 * v], hi[2 * v + 1]);
    }

    #pragma unroll
    for (int t = 0; t < 8; ++t) {
      Frag vf;  // V^T rows (hd) as A-operand, k = keys
      load_a_frag(vf, vbase + (size_t)(t * 16) * SLEN, SLEN, kc);
      acc[t] = wmma_bf16(vf, pf, acc[t]);
    }
  }

  // O^T tile t: lane holds its query's hd = t*16 + half*8 + r (r=0..7) -> one b128 store
  const float inv = 1.0f / l_i;
  unsigned short* ob = AO + ((size_t)(b * SLEN + q0 + n)) * DMOD + h * HDIM;
  #pragma unroll
  for (int t = 0; t < 8; ++t) {
    uint4 u4;
    u4.x = pack2bf(acc[t][0] * inv, acc[t][1] * inv);
    u4.y = pack2bf(acc[t][2] * inv, acc[t][3] * inv);
    u4.z = pack2bf(acc[t][4] * inv, acc[t][5] * inv);
    u4.w = pack2bf(acc[t][6] * inv, acc[t][7] * inv);
    *(uint4*)(ob + t * 16 + half * 8) = u4;
  }
}

// ---------------------------------------------------------------------------
// Output projection: out[token][d] = sum_o AO[token][o] * Wo[d][o]  (f32 out)
// One wave computes a 16-token x 128-col strip.
// grid = (B*S/16, 4), block = 128
// ---------------------------------------------------------------------------
__global__ __launch_bounds__(128)
void out_proj_kernel(const unsigned short* __restrict__ AO,
                     const unsigned short* __restrict__ Wob,
                     float* __restrict__ out)
{
  const int wave = threadIdx.x >> 5;
  const int lane = threadIdx.x & 31;
  const int n = lane & 15, half = lane >> 4;
  const int m0 = blockIdx.x * 16;
  const int n0 = (blockIdx.y * 4 + wave) * 128;

  const unsigned short* abase = AO + (size_t)m0 * DMOD;

  v8f acc[8];
  {
    v8f z = {};
    #pragma unroll
    for (int j = 0; j < 8; ++j) acc[j] = z;
  }
  for (int kk = 0; kk < DMOD; kk += 32) {
    Frag a;
    load_a_frag(a, abase, DMOD, kk);
    #pragma unroll
    for (int j = 0; j < 8; ++j) {
      Frag bf;
      load_b_frag(bf, Wob + (size_t)(n0 + j * 16) * DMOD, DMOD, kk);
      acc[j] = wmma_bf16(a, bf, acc[j]);
    }
  }
  #pragma unroll
  for (int j = 0; j < 8; ++j) {
    const int col = n0 + j * 16 + n;
    #pragma unroll
    for (int r = 0; r < 8; ++r) {
      const int row = m0 + half * 8 + r;
      out[(size_t)row * DMOD + col] = acc[j][r];
    }
  }
}

// ---------------------------------------------------------------------------
// Workspace layout (bytes)
// ---------------------------------------------------------------------------
#define OFF_XB   (size_t)0          // bf16 x           [4096][2048]   16.78 MB
#define OFF_WQB  (size_t)16777216   // bf16 Wq          [2048][2048]    8.39 MB
#define OFF_WKB  (size_t)25165824   // bf16 Wk          [1024][2048]    4.19 MB
#define OFF_WVB  (size_t)29360128   // bf16 Wv          [1024][2048]    4.19 MB
#define OFF_WOB  (size_t)33554432   // bf16 Wo          [2048][2048]    8.39 MB
#define OFF_Q    (size_t)41943040   // bf16 Q           [2][16][2048][128]
#define OFF_K    (size_t)58720256   // bf16 K           [2][8][2048][128]
#define OFF_VT   (size_t)67108864   // bf16 V^T         [2][8][128][2048]
#define OFF_AO   (size_t)75497472   // bf16 attn-out    [2][2048][2048]
#define WS_NEED  (size_t)92274688

extern "C" void kernel_launch(void* const* d_in, const int* in_sizes, int n_in,
                              void* d_out, int out_size, void* d_ws, size_t ws_size,
                              hipStream_t stream) {
  (void)in_sizes; (void)n_in; (void)out_size;
  if (ws_size < WS_NEED) return;

  const float* x   = (const float*)d_in[0];
  const float* Wq  = (const float*)d_in[1];
  const float* Wk  = (const float*)d_in[2];
  const float* Wv  = (const float*)d_in[3];
  const float* Wo  = (const float*)d_in[4];
  const float* qnw = (const float*)d_in[5];
  const float* knw = (const float*)d_in[6];

  char* ws = (char*)d_ws;
  unsigned short* Xb  = (unsigned short*)(ws + OFF_XB);
  unsigned short* Wqb = (unsigned short*)(ws + OFF_WQB);
  unsigned short* Wkb = (unsigned short*)(ws + OFF_WKB);
  unsigned short* Wvb = (unsigned short*)(ws + OFF_WVB);
  unsigned short* Wob = (unsigned short*)(ws + OFF_WOB);
  unsigned short* Qw  = (unsigned short*)(ws + OFF_Q);
  unsigned short* Kw  = (unsigned short*)(ws + OFF_K);
  unsigned short* Vtw = (unsigned short*)(ws + OFF_VT);
  unsigned short* AOw = (unsigned short*)(ws + OFF_AO);

  // 1) f32 -> bf16 conversions (weights become L2-resident bf16)
  cvt_f32_to_bf16<<<dim3(16384), dim3(256), 0, stream>>>(x,  (unsigned int*)Xb,  4194304);
  cvt_f32_to_bf16<<<dim3(8192),  dim3(256), 0, stream>>>(Wq, (unsigned int*)Wqb, 2097152);
  cvt_f32_to_bf16<<<dim3(4096),  dim3(256), 0, stream>>>(Wk, (unsigned int*)Wkb, 1048576);
  cvt_f32_to_bf16<<<dim3(4096),  dim3(256), 0, stream>>>(Wv, (unsigned int*)Wvb, 1048576);
  cvt_f32_to_bf16<<<dim3(8192),  dim3(256), 0, stream>>>(Wo, (unsigned int*)Wob, 2097152);

  // 2) fused QKV projection + RMSNorm + RoPE (+ V transpose)
  qkv_proj_kernel<<<dim3(SLEN / 16, NH + 2 * NKVH, BSZ), dim3(32), 0, stream>>>(
      Xb, Wqb, Wkb, Wvb, qnw, knw, Qw, Kw, Vtw);

  // 3) causal flash attention
  flash_attn_kernel<<<dim3(SLEN / 64, NH, BSZ), dim3(128), 0, stream>>>(Qw, Kw, Vtw, AOw);

  // 4) output projection (f32)
  out_proj_kernel<<<dim3(BSZ * SLEN / 16, 4, 1), dim3(128), 0, stream>>>(AOw, Wob, (float*)d_out);
}
